// GQAAttention_65859028517013
// MI455X (gfx1250) — matrix operand order repaired and checked by reading, not compiled, not run on hardware
//
#include <hip/hip_runtime.h>
#include <hip/hip_bf16.h>

// ---------------------------------------------------------------------------
// GQA attention forward for MI455X (gfx1250, wave32, WMMA).
// B=2, T=2048, D=2048, H=32, d=64, G=8 (GQA group size 4).
// f16 matrix inputs, f32 WMMA accumulation throughout.
// ---------------------------------------------------------------------------

typedef __attribute__((ext_vector_type(16))) _Float16 v16h;
typedef __attribute__((ext_vector_type(8)))  _Float16 v8h;
typedef __attribute__((ext_vector_type(8)))  float    v8f;

#define WMMA_F16(a, b, c) \
  __builtin_amdgcn_wmma_f32_16x16x32_f16(false, (a), false, (b), (short)0, (c), false, false)

#define CAT16(lo, hi) \
  __builtin_shufflevector((lo), (hi), 0,1,2,3,4,5,6,7,8,9,10,11,12,13,14,15)

__device__ __forceinline__ float hmax16(float v) {
  v = fmaxf(v, __shfl_xor(v, 1, 32));
  v = fmaxf(v, __shfl_xor(v, 2, 32));
  v = fmaxf(v, __shfl_xor(v, 4, 32));
  v = fmaxf(v, __shfl_xor(v, 8, 32));
  return v;
}
__device__ __forceinline__ float hsum16(float v) {
  v += __shfl_xor(v, 1, 32);
  v += __shfl_xor(v, 2, 32);
  v += __shfl_xor(v, 4, 32);
  v += __shfl_xor(v, 8, 32);
  return v;
}
__device__ __forceinline__ v8f vzero8() {
  v8f z = {0.f, 0.f, 0.f, 0.f, 0.f, 0.f, 0.f, 0.f};
  return z;
}

// ---------------------------------------------------------------------------
// fp32 -> f16 conversion
// ---------------------------------------------------------------------------
__global__ void cvt_f32_f16(const float* __restrict__ in, _Float16* __restrict__ out, int n) {
  int i = blockIdx.x * blockDim.x + threadIdx.x;
  if (i < n) out[i] = (_Float16)in[i];
}

// ---------------------------------------------------------------------------
// Generic GEMM: Y[M=4096, N] = A[M, K](f16, row-major) * W[N, K](f16)^T + bias
// Wave tile: 32 (M) x 64 (N). 8 waves / block.
// mode 0: store f16 row-major [M, N]
// mode 1: store f16 "V transposed": out[(b*512 + n) * 2048 + (m % 2048)]
// mode 2: store f32 row-major [M, N]
// ---------------------------------------------------------------------------
__global__ void __launch_bounds__(256) proj_gemm(
    const _Float16* __restrict__ A, const _Float16* __restrict__ W,
    const float* __restrict__ bias, _Float16* __restrict__ outH,
    float* __restrict__ outF, int N, int K, int mode)
{
  const int lane = threadIdx.x & 31;
  const int wid  = threadIdx.x >> 5;
  const int l16  = lane & 15;
  const int hh   = lane >> 4;     // half-wave index
  const int ntn  = N >> 6;
  const int wv   = blockIdx.x * 8 + wid;
  const int m0   = (wv / ntn) * 32;
  const int n0   = (wv % ntn) * 64;

  v8f c[2][4];
#pragma unroll
  for (int i = 0; i < 2; i++)
#pragma unroll
    for (int j = 0; j < 4; j++) c[i][j] = vzero8();

  for (int k0 = 0; k0 < K; k0 += 32) {
    // A fragments: lane row = m0+mi*16+l16; K=[k0+16*hh,+16) contiguous
    v16h a[2];
#pragma unroll
    for (int mi = 0; mi < 2; mi++) {
      const _Float16* ap = A + (size_t)(m0 + mi * 16 + l16) * K + k0 + 16 * hh;
      v8h lo = *(const v8h*)ap;
      v8h hi = *(const v8h*)(ap + 8);
      a[mi] = CAT16(lo, hi);
    }
    // B fragments: lane col n = n0+ni*16+l16; K=[k0+16*hh,+16) contiguous in W row n
#pragma unroll
    for (int ni = 0; ni < 4; ni++) {
      const _Float16* wp = W + (size_t)(n0 + ni * 16 + l16) * K + k0 + 16 * hh;
      v16h bf = *(const v16h*)wp;
      c[0][ni] = WMMA_F16(a[0], bf, c[0][ni]);
      c[1][ni] = WMMA_F16(a[1], bf, c[1][ni]);
    }
  }

  float bv[4];
#pragma unroll
  for (int ni = 0; ni < 4; ni++) bv[ni] = bias[n0 + ni * 16 + l16];

#pragma unroll
  for (int mi = 0; mi < 2; mi++)
#pragma unroll
    for (int ni = 0; ni < 4; ni++)
#pragma unroll
      for (int p = 0; p < 8; p++) {
        float v = c[mi][ni][p] + bv[ni];
        int m = m0 + mi * 16 + p + 8 * hh;     // C layout: row = p (+8 for upper half)
        int n = n0 + ni * 16 + l16;            //           col = lane % 16
        if (mode == 0) {
          outH[(size_t)m * N + n] = (_Float16)v;
        } else if (mode == 2) {
          outF[(size_t)m * N + n] = v;
        } else {  // mode 1: transposed V store, rows = (b*512 + n), cols = t
          int bb = m >> 11, t = m & 2047;
          outH[((size_t)(bb * 512 + n)) * 2048 + t] = (_Float16)v;
        }
      }
}

// ---------------------------------------------------------------------------
// Flash attention: one wave = one (b, h, 16-query block), full d=64 accumulator.
// Qh: [B*T, 2048] f16; Kh: [B*T, 512] f16; Vt: [B*512, 2048] f16 (transposed);
// Ah out: [B*T, 2048] f16.
// ---------------------------------------------------------------------------
__global__ void __launch_bounds__(128) attn_kernel(
    const _Float16* __restrict__ Qh, const _Float16* __restrict__ Kh,
    const _Float16* __restrict__ Vt, const float* __restrict__ amask,
    _Float16* __restrict__ Ah)
{
  __shared__ _Float16 plds[4 * 512];   // 1 KB per wave: P tile in A-fragment order

  const int lane = threadIdx.x & 31;
  const int wid  = threadIdx.x >> 5;
  const int l16  = lane & 15;
  const int hh   = lane >> 4;
  const int wv   = blockIdx.x * 4 + wid;
  const int qblk = wv & 127;
  const int h    = (wv >> 7) & 31;
  const int b    = wv >> 12;
  const int qb   = qblk * 16;
  const int g    = h >> 2;             // KV head (group size 4)

  // Q fragments (A-layout): rows qb..qb+15, dims [0,32) and [32,64)
  const _Float16* qp = Qh + (size_t)(b * 2048 + qb + l16) * 2048 + h * 64;
  v16h a0, a1;
  {
    v8h lo0 = *(const v8h*)(qp + 16 * hh);
    v8h hi0 = *(const v8h*)(qp + 8 + 16 * hh);
    a0 = CAT16(lo0, hi0);
    v8h lo1 = *(const v8h*)(qp + 32 + 16 * hh);
    v8h hi1 = *(const v8h*)(qp + 40 + 16 * hh);
    a1 = CAT16(lo1, hi1);
  }

  float m[8], l[8];
#pragma unroll
  for (int p = 0; p < 8; p++) { m[p] = -3.0e38f; l[p] = 0.f; }
  v8f o[4];
#pragma unroll
  for (int f = 0; f < 4; f++) o[f] = vzero8();

  _Float16* myp = plds + wid * 512;

  // causal: only key blocks with kb <= qb+15 contribute
  for (int kb = 0; kb <= qb + 15; kb += 32) {
    // ---- S = Q * K^T (16 x 32), f32 accum ----
    v8f s0 = vzero8(), s1 = vzero8();
    {
      const _Float16* k0p = Kh + (size_t)(b * 2048 + kb + l16) * 512 + g * 64 + 16 * hh;
      v16h b00 = *(const v16h*)k0p;          // dims 0..31
      v16h b01 = *(const v16h*)(k0p + 32);   // dims 32..63
      s0 = WMMA_F16(a0, b00, s0);
      s0 = WMMA_F16(a1, b01, s0);
      const _Float16* k1p = k0p + (size_t)16 * 512;
      v16h b10 = *(const v16h*)k1p;
      v16h b11 = *(const v16h*)(k1p + 32);
      s1 = WMMA_F16(a0, b10, s1);
      s1 = WMMA_F16(a1, b11, s1);
    }

    // ---- scale + masks + streaming softmax ----
    const int j0 = kb + l16, j1 = kb + 16 + l16;
    const float pad0 = (1.f - amask[b * 2048 + j0]) * -1e9f;
    const float pad1 = (1.f - amask[b * 2048 + j1]) * -1e9f;

#pragma unroll
    for (int p = 0; p < 8; p++) {
      int iq = qb + p + 8 * hh;
      float v0 = s0[p] * 0.125f + pad0 + (j0 > iq ? -1e9f : 0.f);
      float v1 = s1[p] * 0.125f + pad1 + (j1 > iq ? -1e9f : 0.f);
      float mx = hmax16(fmaxf(v0, v1));
      float mn = fmaxf(m[p], mx);
      float al = __expf(m[p] - mn);
      m[p] = mn;
      float e0 = __expf(v0 - mn);
      float e1 = __expf(v1 - mn);
      float rs = hsum16(e0 + e1);
      l[p] = l[p] * al + rs;
      o[0][p] *= al; o[1][p] *= al; o[2][p] *= al; o[3][p] *= al;
      // stash P in LDS directly in A-fragment element order:
      // value (r, c): owner lane = r + 16*(c>>4), element = c&15
      int rl = p + 8 * hh;
      int idx = rl * 16 + l16;
      myp[idx]       = (_Float16)e0;   // c = l16
      myp[idx + 256] = (_Float16)e1;   // c = 16 + l16
    }

    // reload P as a contiguous A-fragment (32 B per lane)
    v16h pa = *(const v16h*)(myp + lane * 16);

    // ---- O += P * V ----
#pragma unroll
    for (int f = 0; f < 4; f++) {
      const _Float16* vp =
          Vt + (size_t)(b * 512 + g * 64 + f * 16 + l16) * 2048 + kb + 16 * hh;
      v16h vb = *(const v16h*)vp;
      o[f] = WMMA_F16(pa, vb, o[f]);
    }
  }

  // ---- normalize + store attn output (f16, [B*T, 2048]) ----
#pragma unroll
  for (int f = 0; f < 4; f++)
#pragma unroll
    for (int p = 0; p < 8; p++) {
      float v = o[f][p] / l[p];
      int r = qb + p + 8 * hh;
      Ah[(size_t)(b * 2048 + r) * 2048 + h * 64 + f * 16 + l16] = (_Float16)v;
    }
}

// ---------------------------------------------------------------------------
// Host-side launch
// ---------------------------------------------------------------------------
extern "C" void kernel_launch(void* const* d_in, const int* in_sizes, int n_in,
                              void* d_out, int out_size, void* d_ws, size_t ws_size,
                              hipStream_t stream) {
  (void)in_sizes; (void)n_in; (void)out_size; (void)ws_size;

  const float* X  = (const float*)d_in[0];
  const float* am = (const float*)d_in[1];
  const float* Wq = (const float*)d_in[2];
  const float* bq = (const float*)d_in[3];
  const float* Wk = (const float*)d_in[4];
  const float* bk = (const float*)d_in[5];
  const float* Wv = (const float*)d_in[6];
  const float* bv = (const float*)d_in[7];
  const float* Wo = (const float*)d_in[8];
  const float* bo = (const float*)d_in[9];
  float* out = (float*)d_out;

  char* ws = (char*)d_ws;
  _Float16* Xh  = (_Float16*)(ws + 0);          // 16 MB : X  f16 [4096, 2048]
  _Float16* Wqh = (_Float16*)(ws + 16777216);   //  8 MB : Wq f16 [2048, 2048]
  _Float16* Wkh = (_Float16*)(ws + 25165824);   //  2 MB : Wk f16 [512, 2048]
  _Float16* Wvh = (_Float16*)(ws + 27262976);   //  2 MB : Wv f16 [512, 2048]
  _Float16* Woh = (_Float16*)(ws + 29360128);   //  8 MB : Wo f16 [2048, 2048]
  _Float16* Qh  = (_Float16*)(ws + 37748736);   // 16 MB : Q  f16 [4096, 2048]
  _Float16* Kh  = (_Float16*)(ws + 54525952);   //  4 MB : K  f16 [4096, 512]
  _Float16* Vt  = (_Float16*)(ws + 58720256);   //  4 MB : V^T f16 [1024, 2048]
  _Float16* Ah  = (_Float16*)(ws + 62914560);   // 16 MB : attn out f16 [4096, 2048]

  // fp32 -> f16 conversions
  cvt_f32_f16<<<(4096 * 2048 + 255) / 256, 256, 0, stream>>>(X,  Xh,  4096 * 2048);
  cvt_f32_f16<<<(2048 * 2048 + 255) / 256, 256, 0, stream>>>(Wq, Wqh, 2048 * 2048);
  cvt_f32_f16<<<( 512 * 2048 + 255) / 256, 256, 0, stream>>>(Wk, Wkh,  512 * 2048);
  cvt_f32_f16<<<( 512 * 2048 + 255) / 256, 256, 0, stream>>>(Wv, Wvh,  512 * 2048);
  cvt_f32_f16<<<(2048 * 2048 + 255) / 256, 256, 0, stream>>>(Wo, Woh, 2048 * 2048);

  // projections: waves = 128 * (N/64), blocks = waves/8
  proj_gemm<<<512, 256, 0, stream>>>(Xh, Wqh, bq, Qh, nullptr, 2048, 2048, 0);
  proj_gemm<<<128, 256, 0, stream>>>(Xh, Wkh, bk, Kh, nullptr,  512, 2048, 0);
  proj_gemm<<<128, 256, 0, stream>>>(Xh, Wvh, bv, Vt, nullptr,  512, 2048, 1);

  // attention: B*H*(T/16) = 8192 waves, 4 waves / block
  attn_kernel<<<2048, 128, 0, stream>>>(Qh, Kh, Vt, am, Ah);

  // output projection -> fp32 d_out
  proj_gemm<<<512, 256, 0, stream>>>(Ah, Woh, bo, nullptr, out, 2048, 2048, 2);
}
